// SingleGraphConvolutionalLayer_20023137534856
// MI455X (gfx1250) — matrix operand order repaired and checked by reading, not compiled, not run on hardware
//
#include <hip/hip_runtime.h>
#include <math.h>

// ---------------------------------------------------------------------------
// Graph conv layer for MI455X (gfx1250, wave32).
// Dominant cost: streaming A (268 MB) twice -> ~23 us HBM floor @ 23.3 TB/s.
// Message-passing GEMM done with v_wmma_f32_16x16x32_bf16 (f32 accumulate).
// ---------------------------------------------------------------------------

#define GN 2048   // nodes
#define GB 16     // batch
#define GF 128    // features (F_in == H)
#define GC 32     // classes

typedef __attribute__((ext_vector_type(16))) __bf16 v16bf;
typedef __attribute__((ext_vector_type(8)))  float  v8f;
typedef __attribute__((ext_vector_type(4)))  float  f32x4;
typedef __attribute__((ext_vector_type(4)))  unsigned int u32x4;
typedef __attribute__((ext_vector_type(2)))  unsigned int u32x2;

union FragBF { u32x4 q[2]; v16bf v; };

__device__ __forceinline__ unsigned short f2bf(float f) {
    unsigned u = __float_as_uint(f);
    u += 0x7FFFu + ((u >> 16) & 1u);      // round-to-nearest-even
    return (unsigned short)(u >> 16);
}
__device__ __forceinline__ float bf2f(unsigned short h) {
    return __uint_as_float(((unsigned)h) << 16);
}

// ---------------------------------------------------------------------------
// K1: r[b,n] = 1/sqrt(1 + sum_m A[b,n,m]) ; one 256-thread block per row.
// Non-temporal: A is a one-shot stream, keep it out of L2.
// ---------------------------------------------------------------------------
__global__ __launch_bounds__(256) void k1_rowdeg(const float* __restrict__ A,
                                                 float* __restrict__ r) {
    __shared__ float red[256];
    const size_t row = blockIdx.x;                 // b*GN + n
    const float* p = A + row * (size_t)GN;
    const int t = threadIdx.x;
    f32x4 v0 = __builtin_nontemporal_load((const f32x4*)(p + t * 4));
    f32x4 v1 = __builtin_nontemporal_load((const f32x4*)(p + 1024 + t * 4));
    red[t] = (v0.x + v0.y + v0.z + v0.w) + (v1.x + v1.y + v1.z + v1.w);
    __syncthreads();
    for (int off = 128; off > 0; off >>= 1) {
        if (t < off) red[t] += red[t + off];
        __syncthreads();
    }
    if (t == 0) r[row] = rsqrtf(red[0] + 1.0f);
}

// ---------------------------------------------------------------------------
// K2: y = r * leaky(LN(x @ W1 + b1)) -> bf16, stored TRANSPOSED Yt[b][d][m]
// (so the WMMA B-operand is contiguous along K). Block = 128 thr, 32 rows.
// ---------------------------------------------------------------------------
__global__ __launch_bounds__(128) void k2_fc1(const float* __restrict__ x,
                                              const float* __restrict__ W1,
                                              const float* __restrict__ b1,
                                              const float* __restrict__ r,
                                              unsigned short* __restrict__ Yt) {
    __shared__ float xs[32 * 129];                 // padded: bank-conflict-free stats
    __shared__ unsigned short os[128 * 40];        // transposed bf16 tile
    __shared__ float mu_s[32], rs_s[32];
    const int b = blockIdx.y, m0 = blockIdx.x * 32;
    const int t = threadIdx.x;                     // t == output feature d
    const float* gx = x + ((size_t)b * GN + m0) * GF;
    for (int i = 0; i < 32; ++i) {                 // coalesced row loads
        int c = t + 128 * i, m = c >> 7, f = c & 127;
        xs[m * 129 + f] = gx[m * GF + f];
    }
    __syncthreads();
    float acc[32];
#pragma unroll
    for (int m = 0; m < 32; ++m) acc[m] = b1[t];
    for (int f = 0; f < GF; ++f) {
        float w = W1[f * GF + t];                  // coalesced, W1 hot in L2
#pragma unroll
        for (int m = 0; m < 32; ++m) acc[m] = fmaf(xs[m * 129 + f], w, acc[m]);
    }
    __syncthreads();
#pragma unroll
    for (int m = 0; m < 32; ++m) xs[m * 129 + t] = acc[m];
    __syncthreads();
    if (t < 32) {                                  // per-row LN stats
        float s = 0.f, ss = 0.f;
        for (int f = 0; f < GF; ++f) { float v = xs[t * 129 + f]; s += v; ss += v * v; }
        float mu = s * (1.f / GF);
        mu_s[t] = mu;
        rs_s[t] = rsqrtf(ss * (1.f / GF) - mu * mu + 1e-5f);
    }
    __syncthreads();
#pragma unroll
    for (int m = 0; m < 32; ++m) {
        float v = (acc[m] - mu_s[m]) * rs_s[m];
        v = v >= 0.f ? v : 0.02f * v;
        v *= r[(size_t)b * GN + m0 + m];           // fold column scale r[m] in now
        os[t * 40 + m] = f2bf(v);
    }
    __syncthreads();
    unsigned short* gy = Yt + (size_t)b * GF * GN;
#pragma unroll
    for (int i = 0; i < 4; ++i) {                  // coalesced 16B transposed writes
        int c = t + 128 * i, d = c >> 2, q = (c & 3) * 8;
        *(u32x4*)(gy + (size_t)d * GN + m0 + q) = *(const u32x4*)&os[d * 40 + q];
    }
}

// ---------------------------------------------------------------------------
// K3: H[b,n,d] = r[n] * ( (A_bf16 @ y)[n,d] + y[n,d] )   (E2 = A + I, E = 1)
// 256 thr = 8 waves; block tile 128(M) x 128(D); K stepped by 32.
// Wave w -> rows w*16..w*16+15, all 8 D-tiles: batch-issue all 18 fragment
// ds_loads, one wait, then 8 back-to-back v_wmma_f32_16x16x32_bf16.
// ---------------------------------------------------------------------------
__global__ __launch_bounds__(256) void k3_msgpass(const float* __restrict__ A,
                                                  const unsigned short* __restrict__ Yt,
                                                  const float* __restrict__ r,
                                                  float* __restrict__ H) {
    __shared__ unsigned short As[128 * 40];        // 128 rows x 32 K (bf16), pad 8
    __shared__ unsigned short Ys[128 * 40];        // 128 D  x 32 K (bf16), pad 8
    const int b = blockIdx.y, bt = blockIdx.x;
    const int t = threadIdx.x;
    const int wv = t >> 5, lane = t & 31;
    const int hl = lane >> 4, l16 = lane & 15;
    const float* gA = A + ((size_t)b * GN + bt * 128) * GN;
    const unsigned short* gY = Yt + (size_t)b * GF * GN;

    v8f acc[8];
#pragma unroll
    for (int i = 0; i < 8; ++i) { v8f z = {0.f,0.f,0.f,0.f,0.f,0.f,0.f,0.f}; acc[i] = z; }

    for (int k0 = 0; k0 < GN; k0 += 32) {
        // ---- stage A tile: 128x32 f32 -> bf16 in LDS (non-temporal stream) ----
#pragma unroll
        for (int i = 0; i < 4; ++i) {
            int c = t + 256 * i;                   // 0..1023
            int row = c >> 3, col4 = (c & 7) * 4;
            const float* src = gA + (size_t)row * GN + k0 + col4;
            f32x4 v = __builtin_nontemporal_load((const f32x4*)src);
            __builtin_prefetch(src + 32, 0, 0);    // speculative: safe past end
            union { __bf16 h[4]; u32x2 q; } pk;    // v_cvt_pk_bf16_f32 path
            pk.h[0] = (__bf16)v.x; pk.h[1] = (__bf16)v.y;
            pk.h[2] = (__bf16)v.z; pk.h[3] = (__bf16)v.w;
            *(u32x2*)&As[row * 40 + col4] = pk.q;
        }
        // ---- stage Y tile: 128 D-rows x 32 K bf16 (already transposed) ----
#pragma unroll
        for (int i = 0; i < 2; ++i) {
            int c = t + 256 * i;                   // 0..511
            int d = c >> 2, q = (c & 3) * 8;
            *(u32x4*)&Ys[d * 40 + q] = *(const u32x4*)(gY + (size_t)d * GN + k0 + q);
        }
        __syncthreads();

        // ---- A-frag per ISA 16x32 bf16 layout (K pairs per lane) ----
        const int m = wv * 16 + l16;
        const int kbA = hl ? 8 : 0;                // lanes 16-31: K 8..15 / 24..31
        FragBF af;
        af.q[0] = *(const u32x4*)&As[m * 40 + kbA];
        af.q[1] = *(const u32x4*)&As[m * 40 + kbA + 16];
        // ---- preload ALL B-frags (16 ds_load_b128 in flight) ----
        const int kbB = hl ? 8 : 0;                // B: lanes 16-31 hold K 8..15 / 24..31
        FragBF bf[8];
#pragma unroll
        for (int nt = 0; nt < 8; ++nt) {
            const int d = nt * 16 + l16;           // B column = lane%16
            bf[nt].q[0] = *(const u32x4*)&Ys[d * 40 + kbB];
            bf[nt].q[1] = *(const u32x4*)&Ys[d * 40 + kbB + 16];
        }
        // Scheduling fence: keep the ds_loads batched above (one s_wait_dscnt)
        // instead of letting the scheduler sink them between the WMMAs.
        asm volatile("" ::: "memory");
        // ---- 8 WMMAs back-to-back on independent accumulators ----
#pragma unroll
        for (int nt = 0; nt < 8; ++nt)
            acc[nt] = __builtin_amdgcn_wmma_f32_16x16x32_bf16(
                false, af.v, false, bf[nt].v, (short)0, acc[nt], false, false);
        __syncthreads();
    }

    // ---- epilogue: + identity (y[n]) then scale by r[n]; C/D layout:
    //      lane<16 -> M=j, lane>=16 -> M=j+8; N = lane%16 ----
    const int rowBase = bt * 128 + wv * 16 + hl * 8;
    float rv[8];
#pragma unroll
    for (int j = 0; j < 8; ++j) rv[j] = r[(size_t)b * GN + rowBase + j];
#pragma unroll
    for (int nt = 0; nt < 8; ++nt) {
        const int d = nt * 16 + l16;
#pragma unroll
        for (int j = 0; j < 8; ++j) {
            const int row = rowBase + j;
            float yid = bf2f(gY[(size_t)d * GN + row]);   // I * y term
            H[((size_t)b * GN + row) * GF + d] = rv[j] * (acc[nt][j] + yid);
        }
    }
}

// ---------------------------------------------------------------------------
// K4: h2 = leaky(LN(H @ W2 + b2)); per-block max over its 32 rows -> pmax.
// ---------------------------------------------------------------------------
__global__ __launch_bounds__(128) void k4_fc2max(const float* __restrict__ H,
                                                 const float* __restrict__ W2,
                                                 const float* __restrict__ b2,
                                                 float* __restrict__ pmax) {
    __shared__ float xs[32 * 129];
    __shared__ float mu_s[32], rs_s[32];
    const int b = blockIdx.y, m0 = blockIdx.x * 32;
    const int t = threadIdx.x;
    const float* gh = H + ((size_t)b * GN + m0) * GF;
    for (int i = 0; i < 32; ++i) {
        int c = t + 128 * i, m = c >> 7, f = c & 127;
        xs[m * 129 + f] = gh[m * GF + f];
    }
    __syncthreads();
    float acc[32];
#pragma unroll
    for (int m = 0; m < 32; ++m) acc[m] = b2[t];
    for (int f = 0; f < GF; ++f) {
        float w = W2[f * GF + t];
#pragma unroll
        for (int m = 0; m < 32; ++m) acc[m] = fmaf(xs[m * 129 + f], w, acc[m]);
    }
    __syncthreads();
#pragma unroll
    for (int m = 0; m < 32; ++m) xs[m * 129 + t] = acc[m];
    __syncthreads();
    if (t < 32) {
        float s = 0.f, ss = 0.f;
        for (int f = 0; f < GF; ++f) { float v = xs[t * 129 + f]; s += v; ss += v * v; }
        float mu = s * (1.f / GF);
        mu_s[t] = mu;
        rs_s[t] = rsqrtf(ss * (1.f / GF) - mu * mu + 1e-5f);
    }
    __syncthreads();
    float mx = -3.4e38f;
#pragma unroll
    for (int m = 0; m < 32; ++m) {
        float v = (acc[m] - mu_s[m]) * rs_s[m];
        v = v >= 0.f ? v : 0.02f * v;
        mx = fmaxf(mx, v);
    }
    pmax[((size_t)b * 64 + blockIdx.x) * GF + t] = mx;
}

// ---------------------------------------------------------------------------
// K5: hp = max over partials; out = log_softmax(hp @ W3 + b3). One block/batch.
// ---------------------------------------------------------------------------
__global__ __launch_bounds__(128) void k5_head(const float* __restrict__ pmax,
                                               const float* __restrict__ W3,
                                               const float* __restrict__ b3,
                                               float* __restrict__ out) {
    __shared__ float hs[GF];
    __shared__ float ov[GC];
    const int b = blockIdx.x, t = threadIdx.x;
    float mx = -3.4e38f;
    for (int i = 0; i < 64; ++i) mx = fmaxf(mx, pmax[((size_t)b * 64 + i) * GF + t]);
    hs[t] = mx;
    __syncthreads();
    if (t < GC) {
        float s = b3[t];
        for (int d = 0; d < GF; ++d) s = fmaf(hs[d], W3[d * GC + t], s);
        ov[t] = s;
    }
    __syncthreads();
    if (t < GC) {
        float m = -3.4e38f;
        for (int c = 0; c < GC; ++c) m = fmaxf(m, ov[c]);
        float se = 0.f;
        for (int c = 0; c < GC; ++c) se += expf(ov[c] - m);
        out[b * GC + t] = ov[t] - m - logf(se);
    }
}

// ---------------------------------------------------------------------------
extern "C" void kernel_launch(void* const* d_in, const int* in_sizes, int n_in,
                              void* d_out, int out_size, void* d_ws, size_t ws_size,
                              hipStream_t stream) {
    const float* x  = (const float*)d_in[0];
    const float* A  = (const float*)d_in[1];   // [B,1,N,N]
    const float* W1 = (const float*)d_in[2];
    const float* b1 = (const float*)d_in[3];
    const float* W2 = (const float*)d_in[4];
    const float* b2 = (const float*)d_in[5];
    const float* W3 = (const float*)d_in[6];
    const float* b3 = (const float*)d_in[7];
    float* outp = (float*)d_out;

    // workspace layout
    char* ws = (char*)d_ws;
    float*          r    = (float*)ws;                               // 16*2048*4      = 128 KB
    unsigned short* Yt   = (unsigned short*)(ws + 131072);           // 16*128*2048*2  = 8 MB
    float*          H    = (float*)(ws + 131072 + 8388608);          // 16*2048*128*4  = 16 MB
    float*          pmax = (float*)(ws + 131072 + 8388608 + 16777216); // 16*64*128*4  = 512 KB
    (void)in_sizes; (void)n_in; (void)out_size; (void)ws_size;

    k1_rowdeg <<<GB * GN,            256, 0, stream>>>(A, r);
    k2_fc1    <<<dim3(GN / 32, GB),  128, 0, stream>>>(x, W1, b1, r, Yt);
    k3_msgpass<<<dim3(GN / 128, GB), 256, 0, stream>>>(A, Yt, r, H);
    k4_fc2max <<<dim3(GN / 32, GB),  128, 0, stream>>>(H, W2, b2, pmax);
    k5_head   <<<GB,                 128, 0, stream>>>(pmax, W3, b3, outp);
}
